// SlideAttention_42898133353031
// MI455X (gfx1250) — matrix operand order repaired and checked by reading, not compiled, not run on hardware
//
#include <hip/hip_runtime.h>

typedef __attribute__((ext_vector_type(16))) _Float16 v16h;
typedef __attribute__((ext_vector_type(8)))  float    v8f;

union FragA { v16h h; unsigned int u[8]; };

#define BB   8
#define NH   12
#define HD   32
#define NN   1024
#define DIMC 384
#define QKVN 1152

// ---------------- conversion kernels ----------------
__global__ void cvt_x_kernel(const float* __restrict__ x, _Float16* __restrict__ xh, int n) {
    int i = blockIdx.x * blockDim.x + threadIdx.x;
    if (i < n) xh[i] = (_Float16)x[i];
}

// w [K,N] row-major -> wt [N,K] f16
__global__ void transpose_w_kernel(const float* __restrict__ w, _Float16* __restrict__ wt,
                                   int K, int N) {
    int i = blockIdx.x * blockDim.x + threadIdx.x;
    if (i < K * N) {
        int k = i / N, n = i % N;
        wt[(size_t)n * K + k] = (_Float16)w[i];
    }
}

// ---------------- WMMA GEMM: C[M,N] = A[M,K] * Bt[N,K]^T + bias ----------------
// block = 256 threads = 8 waves; block tile 64(M) x 32(N); wave tile 16x16
__global__ __launch_bounds__(256) void gemm_f16_kernel(
    const _Float16* __restrict__ A, const _Float16* __restrict__ Bt,
    const float* __restrict__ bias, float* __restrict__ C,
    int M, int N, int K)
{
    int wave = threadIdx.x >> 5, lane = threadIdx.x & 31;
    int lq = lane & 15, hg = lane >> 4;
    int m0 = blockIdx.y * 64 + (wave & 3) * 16;
    int n0 = blockIdx.x * 32 + (wave >> 2) * 16;

    v8f acc = {};
    const _Float16* ap = A  + (size_t)(m0 + lq) * K + hg * 16;  // A-frag base (K half-offset folded)
    const _Float16* bp = Bt + (size_t)(n0 + lq) * K + hg * 16;  // B-frag base

    for (int k0 = 0; k0 < K; k0 += 32) {
        FragA fa, fb;
#pragma unroll
        for (int v = 0; v < 8; v++) {
            fa.u[v] = *(const unsigned int*)(ap + k0 + v * 2);
            fb.u[v] = *(const unsigned int*)(bp + k0 + v * 2);
        }
        acc = __builtin_amdgcn_wmma_f32_16x16x32_f16(false, fa.h, false, fb.h,
                                                     (short)0, acc, false, false);
    }
#pragma unroll
    for (int r = 0; r < 8; r++) {
        int row = m0 + r + 8 * hg;
        int col = n0 + lq;
        C[(size_t)row * N + col] = acc[r] + bias[col];
    }
}

// ---------------- depthwise conv + split/scale/layout ----------------
// qkv fp32 [B,N,3,NH,HD]  ->  qh [B*NH,N,HD] (q*scale, f16)
//                             kh [B*NH,N,HD] (conv(k), f16)
//                             vhT[B*NH,HD,N] (conv(v), f16, key-major)
__global__ void conv_split_kernel(const float* __restrict__ qkv,
                                  const float* __restrict__ conv_w,
                                  const float* __restrict__ conv_b,
                                  _Float16* __restrict__ qh,
                                  _Float16* __restrict__ kh,
                                  _Float16* __restrict__ vhT)
{
    int idx = blockIdx.x * blockDim.x + threadIdx.x;
    if (idx >= BB * NH * NN * HD) return;
    int d  = idx & 31;
    int n  = (idx >> 5) & 1023;
    int bh = idx >> 15;               // b*NH + h
    int h  = bh % NH, b = bh / NH;
    int x  = n & 31, y = n >> 5;

    const float scale = 0.17677669529663687f; // 32^-0.5
    // qkv element (b,n,c,h,d) at ((b*NN+n)*36 + c*12 + h)*32 + d
    size_t rowb = ((size_t)b * NN + n) * 36;
    qh[((size_t)bh * NN + n) * HD + d] = (_Float16)(qkv[(rowb + h) * 32 + d] * scale);

    float accK = conv_b[d], accV = conv_b[d];
#pragma unroll
    for (int dy = -1; dy <= 1; dy++) {
        int yy = y + dy; if (yy < 0 || yy > 31) continue;
#pragma unroll
        for (int dx = -1; dx <= 1; dx++) {
            int xx = x + dx; if (xx < 0 || xx > 31) continue;
            int nn = yy * 32 + xx;
            float w = conv_w[d * 9 + (dy + 1) * 3 + (dx + 1)];
            size_t base = (((size_t)b * NN + nn) * 36 + 12 + h) * 32 + d; // c=1 (k)
            accK += qkv[base] * w;
            accV += qkv[base + 12 * 32] * w;                             // c=2 (v)
        }
    }
    kh [((size_t)bh * NN + n) * HD + d]  = (_Float16)accK;
    vhT[((size_t)bh * HD + d) * NN + n]  = (_Float16)accV;
}

// ---------------- flash attention ----------------
// grid (96, 16), block 128 (4 waves). Each wave: 16 query rows, full 1024-key sweep.
__global__ __launch_bounds__(128) void attn_kernel(
    const _Float16* __restrict__ qh,    // [B*NH, N, HD], pre-scaled
    const _Float16* __restrict__ kh,    // [B*NH, N, HD]
    const _Float16* __restrict__ vhT,   // [B*NH, HD, N]
    const float*    __restrict__ bias,  // [NH, N, N]
    _Float16*       __restrict__ oh)    // [B*N, DIMC]
{
    int bh   = blockIdx.x;
    int b    = bh / NH, h = bh % NH;
    int wave = threadIdx.x >> 5, lane = threadIdx.x & 31;
    int lq   = lane & 15, hg = lane >> 4;
    int m0   = blockIdx.y * 64 + wave * 16;

    __shared__ _Float16 plds_all[4][16 * 34];
    _Float16* plds = plds_all[wave];

    // Q A-fragment: loaded once, reused for all 64 key blocks
    FragA fq;
    const _Float16* qp = qh + ((size_t)bh * NN + m0 + lq) * HD;
#pragma unroll
    for (int v = 0; v < 8; v++)
        fq.u[v] = *(const unsigned int*)(qp + hg * 16 + v * 2);

    v8f o0 = {}, o1 = {};
    float m_r[8], l_r[8];
#pragma unroll
    for (int r = 0; r < 8; r++) { m_r[r] = -1e30f; l_r[r] = 0.f; }

    const v8f zc = {};
    for (int j = 0; j < NN; j += 32) {
        // K^T B-fragments for key sub-blocks [j..j+15], [j+16..j+31]
        FragA fk0, fk1;
        const _Float16* kp = kh + ((size_t)bh * NN + j + lq) * HD + hg * 16;
#pragma unroll
        for (int v = 0; v < 8; v++) {
            fk0.u[v] = *(const unsigned int*)(kp + v * 2);
            fk1.u[v] = *(const unsigned int*)(kp + 16 * HD + v * 2);
        }
        v8f s0 = __builtin_amdgcn_wmma_f32_16x16x32_f16(false, fq.h, false, fk0.h,
                                                        (short)0, zc, false, false);
        v8f s1 = __builtin_amdgcn_wmma_f32_16x16x32_f16(false, fq.h, false, fk1.h,
                                                        (short)0, zc, false, false);
        // add relative-position bias
#pragma unroll
        for (int r = 0; r < 8; r++) {
            const float* bpr = bias + ((size_t)h * NN + (m0 + r + 8 * hg)) * NN + j + lq;
            s0[r] += bpr[0];
            s1[r] += bpr[16];
        }
        // online softmax (row = 16 lanes of one half-group)
#pragma unroll
        for (int r = 0; r < 8; r++) {
            float mx = fmaxf(s0[r], s1[r]);
#pragma unroll
            for (int off = 1; off < 16; off <<= 1) mx = fmaxf(mx, __shfl_xor(mx, off, 32));
            float mn  = fmaxf(m_r[r], mx);
            float cor = __expf(m_r[r] - mn);
            float p0  = __expf(s0[r] - mn);
            float p1  = __expf(s1[r] - mn);
            float rs  = p0 + p1;
#pragma unroll
            for (int off = 1; off < 16; off <<= 1) rs += __shfl_xor(rs, off, 32);
            l_r[r] = l_r[r] * cor + rs;
            m_r[r] = mn;
            o0[r] *= cor; o1[r] *= cor;
            s0[r] = p0;   s1[r] = p1;
        }
        // P: C-layout -> A-fragment layout via per-wave LDS tile
#pragma unroll
        for (int r = 0; r < 8; r++) {
            int row = r + 8 * hg;
            plds[row * 34 + lq]      = (_Float16)s0[r];
            plds[row * 34 + 16 + lq] = (_Float16)s1[r];
        }
        FragA fp;
#pragma unroll
        for (int v = 0; v < 8; v++)
            fp.u[v] = *(const unsigned int*)(plds + lq * 34 +
                       hg * 16 + v * 2);
        // V B-fragments (dims 0..15 and 16..31)
        FragA fv0, fv1;
        const _Float16* vp = vhT + ((size_t)bh * HD + lq) * NN + j + hg * 16;
#pragma unroll
        for (int v = 0; v < 8; v++) {
            fv0.u[v] = *(const unsigned int*)(vp + v * 2);
            fv1.u[v] = *(const unsigned int*)(vp + (size_t)16 * NN + v * 2);
        }
        o0 = __builtin_amdgcn_wmma_f32_16x16x32_f16(false, fp.h, false, fv0.h,
                                                    (short)0, o0, false, false);
        o1 = __builtin_amdgcn_wmma_f32_16x16x32_f16(false, fp.h, false, fv1.h,
                                                    (short)0, o1, false, false);
    }
    // normalize + store to [B, N, NH*HD] f16 for projection GEMM
#pragma unroll
    for (int r = 0; r < 8; r++) {
        float inv = 1.0f / l_r[r];
        int q = m0 + r + 8 * hg;
        size_t base = ((size_t)b * NN + q) * DIMC + h * HD;
        oh[base + lq]      = (_Float16)(o0[r] * inv);
        oh[base + 16 + lq] = (_Float16)(o1[r] * inv);
    }
}

// ---------------- launcher ----------------
extern "C" void kernel_launch(void* const* d_in, const int* in_sizes, int n_in,
                              void* d_out, int out_size, void* d_ws, size_t ws_size,
                              hipStream_t stream) {
    const float* x          = (const float*)d_in[0];
    const float* w_qkv      = (const float*)d_in[1];
    const float* b_qkv      = (const float*)d_in[2];
    const float* conv_w     = (const float*)d_in[3];
    const float* conv_b     = (const float*)d_in[4];
    const float* w_proj     = (const float*)d_in[5];
    const float* b_proj     = (const float*)d_in[6];
    const float* bias_table = (const float*)d_in[7];
    float* out = (float*)d_out;

    char* ws = (char*)d_ws;
    const size_t M = (size_t)BB * NN;                 // 8192
    _Float16* xh     = (_Float16*)ws;                 ws += M * DIMC * 2;        // 6.29 MB
    _Float16* wqkvT  = (_Float16*)ws;                 ws += (size_t)QKVN * DIMC * 2;
    _Float16* wprojT = (_Float16*)ws;                 ws += (size_t)DIMC * DIMC * 2;
    float*    qkv    = (float*)ws;                    ws += M * QKVN * 4;        // 37.7 MB
    _Float16* qh     = (_Float16*)ws;                 ws += M * DIMC * 2;
    _Float16* kh     = (_Float16*)ws;                 ws += M * DIMC * 2;
    _Float16* vhT    = (_Float16*)ws;                 ws += M * DIMC * 2;
    _Float16* oh     = (_Float16*)ws;                 ws += M * DIMC * 2;

    int nx = (int)(M * DIMC);
    cvt_x_kernel<<<(nx + 255) / 256, 256, 0, stream>>>(x, xh, nx);
    transpose_w_kernel<<<(DIMC * QKVN + 255) / 256, 256, 0, stream>>>(w_qkv, wqkvT, DIMC, QKVN);
    transpose_w_kernel<<<(DIMC * DIMC + 255) / 256, 256, 0, stream>>>(w_proj, wprojT, DIMC, DIMC);

    // QKV GEMM: [8192,1152] = xh[8192,384] @ wqkvT^T + b_qkv
    gemm_f16_kernel<<<dim3(QKVN / 32, M / 64), 256, 0, stream>>>(
        xh, wqkvT, b_qkv, qkv, (int)M, QKVN, DIMC);

    // depthwise conv + split
    int nconv = BB * NH * NN * HD;
    conv_split_kernel<<<(nconv + 255) / 256, 256, 0, stream>>>(
        qkv, conv_w, conv_b, qh, kh, vhT);

    // flash attention: 96 (b,h) x 16 query-blocks of 64
    attn_kernel<<<dim3(BB * NH, NN / 64), 128, 0, stream>>>(
        qh, kh, vhT, bias_table, oh);

    // projection GEMM into d_out
    gemm_f16_kernel<<<dim3(DIMC / 32, M / 64), 256, 0, stream>>>(
        oh, wprojT, b_proj, out, (int)M, DIMC, DIMC);
}
